// GraphSageConv_xe_only_76192719831691
// MI455X (gfx1250) — matrix operand order repaired and checked by reading, not compiled, not run on hardware
//
#include <hip/hip_runtime.h>
#include <hip/hip_bf16.h>

// ---------------------------------------------------------------------------
// GraphSAGE stack for MI455X (gfx1250), wave32 + WMMA bf16.
//   he = segment_sum(edge_feat, edge_dst)                      (fp32 atomics)
//   7x: h = relu(h @ W[:, :K]^T + he * W[:, K] + b)            (WMMA bf16)
//   out = h @ Wr3^T + br3                                      (dot kernel)
// GEMM blocking: 64 nodes x 256 cols per block (8 waves); each wave owns a
// 32-col slab and iterates 4 row tiles, so each B fragment pair feeds 8 WMMAs
// (64 FLOP/byte of weight traffic from L2).
// ---------------------------------------------------------------------------

typedef __attribute__((ext_vector_type(16))) __bf16 v16bf;
typedef __attribute__((ext_vector_type(8)))  __bf16 v8bf;
typedef __attribute__((ext_vector_type(8)))  float  v8f;

#define N_NODES 100000
#define IN_F    128
#define H_F     256
#define MTILE   64       // rows per block

// ---------------------------- small utility kernels ------------------------

__global__ void zero_f32_kernel(float* p, int n) {
    int i = blockIdx.x * blockDim.x + threadIdx.x;
    if (i < n) p[i] = 0.0f;
}

__global__ void scatter_edges_kernel(const float* __restrict__ ef,
                                     const int* __restrict__ dst,
                                     float* __restrict__ he, int n) {
    int i = blockIdx.x * blockDim.x + threadIdx.x;
    if (i < n) unsafeAtomicAdd(&he[dst[i]], ef[i]);
}

// Convert fp32 -> bf16, zero-filling the padded tail [n_valid, n_total).
__global__ void f32_to_bf16_pad_kernel(const float* __restrict__ src,
                                       __bf16* __restrict__ dst,
                                       int n_valid, int n_total) {
    int i = blockIdx.x * blockDim.x + threadIdx.x;
    if (i < n_total) dst[i] = (i < n_valid) ? (__bf16)src[i] : (__bf16)0.0f;
}

// Pack W[rows][src_stride] fp32 -> Wp[rows][K] bf16 (drops the concat column).
__global__ void pack_weight_kernel(const float* __restrict__ src,
                                   __bf16* __restrict__ dst,
                                   int K, int src_stride, int n) {
    int i = blockIdx.x * blockDim.x + threadIdx.x;
    if (i < n) {
        int r = i / K, k = i - r * K;
        dst[i] = (__bf16)src[r * src_stride + k];
    }
}

// out[n] = sum_k h[n][k] * w[k] + b0
__global__ void final_dot_kernel(const __bf16* __restrict__ H,
                                 const float* __restrict__ w,
                                 const float* __restrict__ b,
                                 float* __restrict__ out, int n) {
    int i = blockIdx.x * blockDim.x + threadIdx.x;
    if (i >= n) return;
    const v8bf* hp = (const v8bf*)(H + (size_t)i * H_F);
    float acc = b[0];
#pragma unroll 4
    for (int k = 0; k < H_F / 8; ++k) {
        v8bf h8 = hp[k];
#pragma unroll
        for (int j = 0; j < 8; ++j) acc += (float)h8[j] * w[k * 8 + j];
    }
    out[i] = acc;
}

// ------------------------------- WMMA GEMM layer ---------------------------
// Block: 256 threads (8 waves). Block owns MTILE(64) nodes x 256 columns.
// Wave w: columns [32w, 32w+32) as two 16-col tiles; 4 row tiles of 16.
// A tile (64 x K bf16) staged in LDS once per block, reused by all 8 waves.
template <int K, bool USE_HE>
__global__ __launch_bounds__(256)
void sage_wmma_layer_kernel(const __bf16* __restrict__ X,     // [Npad, K]
                            const __bf16* __restrict__ Wp,    // [256, K] packed
                            const float*  __restrict__ he,    // [Npad]
                            const float*  __restrict__ wlast, // col K of W, stride ws
                            int ws,
                            const float*  __restrict__ bias,  // [256]
                            __bf16* __restrict__ Y) {         // [Npad, 256]
    __shared__ __bf16 As[MTILE * K];

    const int tid   = threadIdx.x;
    const int ntile = blockIdx.x;
    const __bf16* Xt = X + (size_t)ntile * MTILE * K;

    // Cooperative stage of the 64xK A tile into LDS (16B vectors).
#pragma unroll
    for (int i = tid; i < (MTILE * K) / 8; i += 256) {
        ((v8bf*)As)[i] = ((const v8bf*)Xt)[i];
    }
    __syncthreads();

    const int lane = tid & 31;
    const int wave = tid >> 5;
    const int lo16 = lane & 15;
    const int hi   = lane >> 4;           // 0 or 1
    const int col0 = wave * 32 + lo16;    // first tile column for this lane
    const int col1 = col0 + 16;           // second tile column

    v8f c[4][2] = {};

#pragma unroll
    for (int kc = 0; kc < K; kc += 32) {
        // B fragments: lane holds column col, K = kc + hi*16 .. +15 (contiguous
        // in packed row-major W[o][k]).
        const int kb = kc + hi * 16;
        const v16bf b0 = *(const v16bf*)(Wp + (size_t)col0 * K + kb);
        const v16bf b1 = *(const v16bf*)(Wp + (size_t)col1 * K + kb);

#pragma unroll
        for (int r = 0; r < 4; ++r) {
            // A fragment for row tile r: row = r*16 + lo16; lane<16 holds
            // K {kc+0..15}, lane>=16 holds K {kc+16..31}.
            const __bf16* ap = &As[(r * 16 + lo16) * K + kc + hi * 16];
            v8bf alo = *(const v8bf*)ap;
            v8bf ahi = *(const v8bf*)(ap + 8);
            v16bf a;
#pragma unroll
            for (int j = 0; j < 8; ++j) { a[j] = alo[j]; a[j + 8] = ahi[j]; }

            c[r][0] = __builtin_amdgcn_wmma_f32_16x16x32_bf16(
                false, a, false, b0, (short)0, c[r][0], false, false);
            c[r][1] = __builtin_amdgcn_wmma_f32_16x16x32_bf16(
                false, a, false, b1, (short)0, c[r][1], false, false);
        }
    }

    // Epilogue: D layout — VGPR j holds (M = j + 8*hi, N = lo16).
    const float wl0 = USE_HE ? wlast[col0 * ws] : 0.0f;
    const float wl1 = USE_HE ? wlast[col1 * ws] : 0.0f;
    const float b0s = bias[col0];
    const float b1s = bias[col1];

#pragma unroll
    for (int r = 0; r < 4; ++r) {
#pragma unroll
        for (int j = 0; j < 8; ++j) {
            const int nrow = ntile * MTILE + r * 16 + j + hi * 8;
            const float hv = USE_HE ? he[nrow] : 0.0f;
            float v0 = c[r][0][j] + hv * wl0 + b0s;
            float v1 = c[r][1][j] + hv * wl1 + b1s;
            v0 = v0 > 0.0f ? v0 : 0.0f;
            v1 = v1 > 0.0f ? v1 : 0.0f;
            Y[(size_t)nrow * H_F + col0] = (__bf16)v0;
            Y[(size_t)nrow * H_F + col1] = (__bf16)v1;
        }
    }
}

// ------------------------------- host launcher -----------------------------

extern "C" void kernel_launch(void* const* d_in, const int* in_sizes, int n_in,
                              void* d_out, int out_size, void* d_ws, size_t ws_size,
                              hipStream_t stream) {
    (void)n_in; (void)out_size; (void)ws_size;

    const float* node_feat = (const float*)d_in[0];
    const float* edge_feat = (const float*)d_in[1];
    const int*   edge_dst  = (const int*)d_in[2];
    const float* W1  = (const float*)d_in[3];   const float* b1  = (const float*)d_in[4];
    const float* Wm1 = (const float*)d_in[5];   const float* bm1 = (const float*)d_in[6];
    const float* Wm2 = (const float*)d_in[7];   const float* bm2 = (const float*)d_in[8];
    const float* Wm3 = (const float*)d_in[9];   const float* bm3 = (const float*)d_in[10];
    const float* Wm4 = (const float*)d_in[11];  const float* bm4 = (const float*)d_in[12];
    const float* Wr1 = (const float*)d_in[13];  const float* br1 = (const float*)d_in[14];
    const float* Wr2 = (const float*)d_in[15];  const float* br2 = (const float*)d_in[16];
    const float* Wr3 = (const float*)d_in[17];  const float* br3 = (const float*)d_in[18];

    const int n_nodes = in_sizes[0] / IN_F;
    const int n_edges = in_sizes[1];
    const int n_pad   = ((n_nodes + MTILE - 1) / MTILE) * MTILE;   // 100032

    // Workspace carve-up (256B aligned).
    char* ws = (char*)d_ws;
    size_t off = 0;
    auto carve = [&](size_t bytes) -> char* {
        char* p = ws + off;
        off = (off + bytes + 255) & ~(size_t)255;
        return p;
    };
    float*  he  = (float*)carve((size_t)n_pad * sizeof(float));
    __bf16* X0  = (__bf16*)carve((size_t)n_pad * IN_F * 2);
    __bf16* Hs1 = (__bf16*)carve((size_t)n_pad * H_F * 2);
    __bf16* Hs2 = (__bf16*)carve((size_t)n_pad * H_F * 2);
    __bf16* W1p  = (__bf16*)carve((size_t)H_F * IN_F * 2);
    __bf16* Wm1p = (__bf16*)carve((size_t)H_F * H_F * 2);
    __bf16* Wm2p = (__bf16*)carve((size_t)H_F * H_F * 2);
    __bf16* Wm3p = (__bf16*)carve((size_t)H_F * H_F * 2);
    __bf16* Wm4p = (__bf16*)carve((size_t)H_F * H_F * 2);
    __bf16* Wr1p = (__bf16*)carve((size_t)H_F * H_F * 2);
    __bf16* Wr2p = (__bf16*)carve((size_t)H_F * H_F * 2);

    const int BT = 256;

    // 1) Edge aggregation: he = segment_sum(edge_feat, edge_dst), zero-padded.
    zero_f32_kernel<<<(n_pad + BT - 1) / BT, BT, 0, stream>>>(he, n_pad);
    scatter_edges_kernel<<<(n_edges + BT - 1) / BT, BT, 0, stream>>>(
        edge_feat, edge_dst, he, n_edges);

    // 2) Precision conversion / weight packing.
    {
        int nv = n_nodes * IN_F, nt = n_pad * IN_F;
        f32_to_bf16_pad_kernel<<<(nt + BT - 1) / BT, BT, 0, stream>>>(
            node_feat, X0, nv, nt);
    }
    {
        int n = H_F * IN_F;
        pack_weight_kernel<<<(n + BT - 1) / BT, BT, 0, stream>>>(W1, W1p, IN_F, IN_F + 1, n);
        n = H_F * H_F;
        int g = (n + BT - 1) / BT;
        pack_weight_kernel<<<g, BT, 0, stream>>>(Wm1, Wm1p, H_F, H_F + 1, n);
        pack_weight_kernel<<<g, BT, 0, stream>>>(Wm2, Wm2p, H_F, H_F + 1, n);
        pack_weight_kernel<<<g, BT, 0, stream>>>(Wm3, Wm3p, H_F, H_F + 1, n);
        pack_weight_kernel<<<g, BT, 0, stream>>>(Wm4, Wm4p, H_F, H_F + 1, n);
        pack_weight_kernel<<<g, BT, 0, stream>>>(Wr1, Wr1p, H_F, H_F + 1, n);
        pack_weight_kernel<<<g, BT, 0, stream>>>(Wr2, Wr2p, H_F, H_F, n);
    }

    // 3) WMMA layer chain over padded rows.
    const int ntiles = n_pad / MTILE;
    sage_wmma_layer_kernel<IN_F, true><<<ntiles, 256, 0, stream>>>(
        X0, W1p, he, W1 + IN_F, IN_F + 1, b1, Hs1);
    sage_wmma_layer_kernel<H_F, true><<<ntiles, 256, 0, stream>>>(
        Hs1, Wm1p, he, Wm1 + H_F, H_F + 1, bm1, Hs2);
    sage_wmma_layer_kernel<H_F, true><<<ntiles, 256, 0, stream>>>(
        Hs2, Wm2p, he, Wm2 + H_F, H_F + 1, bm2, Hs1);
    sage_wmma_layer_kernel<H_F, true><<<ntiles, 256, 0, stream>>>(
        Hs1, Wm3p, he, Wm3 + H_F, H_F + 1, bm3, Hs2);
    sage_wmma_layer_kernel<H_F, true><<<ntiles, 256, 0, stream>>>(
        Hs2, Wm4p, he, Wm4 + H_F, H_F + 1, bm4, Hs1);
    sage_wmma_layer_kernel<H_F, true><<<ntiles, 256, 0, stream>>>(
        Hs1, Wr1p, he, Wr1 + H_F, H_F + 1, br1, Hs2);
    sage_wmma_layer_kernel<H_F, false><<<ntiles, 256, 0, stream>>>(
        Hs2, Wr2p, nullptr, nullptr, 0, br2, Hs1);

    // 4) Final regression head: [N,256] x [256] + b -> fp32 out.
    final_dot_kernel<<<(n_nodes + BT - 1) / BT, BT, 0, stream>>>(
        Hs1, Wr3, br3, (float*)d_out, n_nodes);
}
